// AtomPairProjection_83416854823431
// MI455X (gfx1250) — matrix operand order repaired and checked by reading, not compiled, not run on hardware
//
#include <hip/hip_runtime.h>
#include <math.h>

// ---- problem constants ----
#define D_ATOM 512
#define D_HID  32
#define D_OUT  128
#define BATCH  4
#define NSEQ   256
#define NROW   (BATCH * NSEQ)   // 1024 (b,i) rows
#define TWO_HID (2 * D_HID)     // 64

typedef _Float16 half8  __attribute__((ext_vector_type(8)));
typedef _Float16 half16 __attribute__((ext_vector_type(16)));
typedef float    float8 __attribute__((ext_vector_type(8)));

// =====================================================================
// K1: ab = z @ W_in^T + b_in ; exact GELU ; LayerNorm(64) ; split a|b
//     one block per (b,i) row, 64 threads = one output column each.
//     Outputs row-major f16: aF[1024][32], bF[1024][32].
// =====================================================================
__global__ __launch_bounds__(TWO_HID)
void k_linear_gelu_ln(const float* __restrict__ z, const float* __restrict__ Win,
                      const float* __restrict__ bin, const float* __restrict__ gamma,
                      const float* __restrict__ beta,
                      _Float16* __restrict__ aF, _Float16* __restrict__ bF) {
  __shared__ float sv[TWO_HID];
  const int row = blockIdx.x;
  const int col = threadIdx.x;
  const float* zr = z + (size_t)row * D_ATOM;
  const float* wr = Win + (size_t)col * D_ATOM;
  float acc = bin[col];
#pragma unroll 4
  for (int k = 0; k < D_ATOM; k += 4) {
    float4 zv = *(const float4*)(zr + k);
    float4 wv = *(const float4*)(wr + k);
    acc += zv.x * wv.x + zv.y * wv.y + zv.z * wv.z + zv.w * wv.w;
  }
  // exact GELU: 0.5*x*(1+erf(x/sqrt(2)))
  const float g = 0.5f * acc * (1.0f + erff(acc * 0.70710678118654752440f));
  sv[col] = g;
  __syncthreads();
  float s = 0.f, s2 = 0.f;
#pragma unroll
  for (int k = 0; k < TWO_HID; ++k) { const float v = sv[k]; s += v; s2 += v * v; }
  const float mu  = s * (1.0f / TWO_HID);
  const float var = s2 * (1.0f / TWO_HID) - mu * mu;
  const float rn  = rsqrtf(var + 1e-5f);
  const float y   = (g - mu) * rn * gamma[col] + beta[col];
  const _Float16 h = (_Float16)y;
  if (col < D_HID) aF[(size_t)row * D_HID + col] = h;
  else             bF[(size_t)row * D_HID + (col - D_HID)] = h;
}

// =====================================================================
// K2: repack W_out[o][c*32+e] into stage-2 B fragments (f16, contiguous-k layout).
//     Column tile ct (0..255) <-> (e = ct>>3, o-range = (ct&7)*16..+15).
//     WB[ct][lane][i] : lane n = o%16 , k = (lane>>4)*16 + i (= c).
// =====================================================================
__global__ __launch_bounds__(256)
void k_pack_wout(const float* __restrict__ Wout, _Float16* __restrict__ WB) {
  const int t    = blockIdx.x * 256 + threadIdx.x;   // 8192 = 256 ct * 32 lanes
  const int ct   = t >> 5;
  const int lane = t & 31;
  const int n    = lane & 15;
  const int kh   = (lane >> 4) * 16;
  const int e    = ct >> 3;
  const int o    = (ct & 7) * 16 + n;
  _Float16* dst  = WB + (size_t)t * 16;
#pragma unroll
  for (int i = 0; i < 16; ++i)
    dst[i] = (_Float16)Wout[(size_t)o * (D_HID * D_HID) + (size_t)(kh + i) * D_HID + e];
}

// =====================================================================
// K3: stage-2 GEMM  t[(bi),(e,o)] = a[1024x32] @ Wbig[32x4096]
//     one v_wmma_f32_16x16x32_f16 per 16x16 tile (K=32 in one shot).
//     D is scattered directly into the B-fragment layout consumed by K4:
//     tB[bi][ot][lane][i] with o = ot*16+(lane&15), e = (lane>>4)*16+i.
// =====================================================================
__global__ __launch_bounds__(256)
void k_phaseB(const _Float16* __restrict__ aF, const _Float16* __restrict__ WB,
              _Float16* __restrict__ tB) {
  const int wave = threadIdx.x >> 5;
  const int lane = threadIdx.x & 31;
  const int tile = blockIdx.x * 8 + wave;  // 16384 tiles = 64 bitiles * 256 ct
  const int bitile = tile >> 8;
  const int ct     = tile & 255;
  const int hlf = lane >> 4;
  const int n   = lane & 15;

  // A fragment: 16x32 f16 (row M = lane&15;
  // lanes 0-15 hold K 0-15, lanes 16-31 hold K 16-31, matching B)
  const _Float16* arow = aF + (size_t)(bitile * 16 + n) * D_HID;
  const half8 lo = *(const half8*)(arow + 16 * hlf);
  const half8 hi = *(const half8*)(arow + 16 * hlf + 8);
  const half16 A = __builtin_shufflevector(lo, hi,
      0, 1, 2, 3, 4, 5, 6, 7, 8, 9, 10, 11, 12, 13, 14, 15);

  // B fragment: pre-packed, contiguous per lane
  const half16 Bm = *(const half16*)(WB + (size_t)(ct * 32 + lane) * 16);

  float8 C = {};
  const float8 D = __builtin_amdgcn_wmma_f32_16x16x32_f16(
      false, A, false, Bm, (short)0, C, false, false);

  // scatter D (M = r + 8*hlf, N = n) into stage-3 fragment layout
  const int ot    = ct & 7;                    // o>>4
  const int laneB = ((ct >> 7) << 4) | n;      // ((e>>4)<<4) | (o&15)
  const int eLow  = (ct >> 3) & 15;            // e & 15
  const int bi0   = bitile * 16 + 8 * hlf;
#pragma unroll
  for (int r = 0; r < 8; ++r) {
    const size_t idx = ((size_t)((bi0 + r) * 8 + ot) * 32 + laneB) * 16 + eLow;
    tB[idx] = (_Float16)D[r];
  }
}

// =====================================================================
// K4: stage-3, one block per (b,i): out[j,o] = bm[b] (256x32) @ t[bi] (32x128) + b_out
//     8 waves; wave w does j-tiles {2w,2w+1} x all 8 o-tiles = 16 WMMAs.
//     Bias folded into the WMMA C operand. 131072 WMMAs total; the 128 MB
//     f32 output stream is the roofline (two 64B-contiguous runs per store).
// =====================================================================
__global__ __launch_bounds__(256)
void k_phaseC(const _Float16* __restrict__ bF, const _Float16* __restrict__ tB,
              const float* __restrict__ bout, float* __restrict__ out) {
  const int bi   = blockIdx.x;        // b*256 + i
  const int b    = bi >> 8;
  const int wave = threadIdx.x >> 5;
  const int lane = threadIdx.x & 31;
  const int hlf  = lane >> 4;
  const int n    = lane & 15;

  half16 Bfrag[8];
  float  bias[8];
#pragma unroll
  for (int ot = 0; ot < 8; ++ot) {
    Bfrag[ot] = *(const half16*)(tB + ((size_t)(bi * 8 + ot) * 32 + lane) * 16);
    bias[ot]  = bout[ot * 16 + n];    // bias depends only on N=o
  }

#pragma unroll
  for (int jj = 0; jj < 2; ++jj) {
    const int jt = wave * 2 + jj;
    const _Float16* brow = bF + ((size_t)b * NSEQ + jt * 16 + n) * D_HID;
    const half8 lo = *(const half8*)(brow + 16 * hlf);
    const half8 hi = *(const half8*)(brow + 16 * hlf + 8);
    const half16 A = __builtin_shufflevector(lo, hi,
        0, 1, 2, 3, 4, 5, 6, 7, 8, 9, 10, 11, 12, 13, 14, 15);
#pragma unroll
    for (int ot = 0; ot < 8; ++ot) {
      float8 C;
#pragma unroll
      for (int k = 0; k < 8; ++k) C[k] = bias[ot];
      const float8 D = __builtin_amdgcn_wmma_f32_16x16x32_f16(
          false, A, false, Bfrag[ot], (short)0, C, false, false);
      float* op = out + ((size_t)bi * NSEQ + jt * 16 + 8 * hlf) * D_OUT + ot * 16 + n;
#pragma unroll
      for (int r = 0; r < 8; ++r) op[(size_t)r * D_OUT] = D[r];  // M = r + 8*hlf
    }
  }
}

// =====================================================================
extern "C" void kernel_launch(void* const* d_in, const int* in_sizes, int n_in,
                              void* d_out, int out_size, void* d_ws, size_t ws_size,
                              hipStream_t stream) {
  const float* z     = (const float*)d_in[0];
  const float* Win   = (const float*)d_in[1];
  const float* bin   = (const float*)d_in[2];
  const float* gamma = (const float*)d_in[3];
  const float* beta  = (const float*)d_in[4];
  const float* Wout  = (const float*)d_in[5];
  const float* bout  = (const float*)d_in[6];
  float* out = (float*)d_out;

  char* ws = (char*)d_ws;
  _Float16* aF = (_Float16*)(ws + 0);         //  64 KB : a  [1024][32] f16
  _Float16* bF = (_Float16*)(ws + 65536);     //  64 KB : b  [1024][32] f16
  _Float16* WB = (_Float16*)(ws + 131072);    // 256 KB : W_out B-fragments
  _Float16* tB = (_Float16*)(ws + 393216);    //   8 MB : t fragments [1024][8][32][16]

  k_linear_gelu_ln<<<NROW, TWO_HID, 0, stream>>>(z, Win, bin, gamma, beta, aF, bF);
  k_pack_wout<<<32, 256, 0, stream>>>(Wout, WB);
  k_phaseB<<<2048, 256, 0, stream>>>(aF, WB, tB);
  k_phaseC<<<NROW, 256, 0, stream>>>(bF, tB, bout, out);
}
